// HopfieldLayer_38216619000478
// MI455X (gfx1250) — matrix operand order repaired and checked by reading, not compiled, not run on hardware
//
#include <hip/hip_runtime.h>

typedef _Float16 v16h __attribute__((ext_vector_type(16)));
typedef float    v8f  __attribute__((ext_vector_type(8)));

namespace {
constexpr float kDT    = 0.1f;
constexpr float kTOL   = 0.1f;
constexpr int   kSteps = 50;
}

// ---------------------------------------------------------------------------
// Setup: build M = -2 W W^T, c = -2 b W^T, and the exact RK4 affine map
//   G = I + (h/2)M + (h^2/6)M^2 + (h^3/24)M^3,  R = I + h*M*G,  s = h*c*G
// Outputs: MT/RT = transposed f16 (column-of-M contiguous) for WMMA B gathers.
// ---------------------------------------------------------------------------
__global__ void hop_setup(const float* __restrict__ W,
                          const float* __restrict__ bias,
                          _Float16* __restrict__ MT,
                          _Float16* __restrict__ RT,
                          float* __restrict__ cOut,
                          float* __restrict__ sOut) {
  __shared__ float Ml[64 * 64];
  __shared__ float Al[64 * 64];
  __shared__ float Bl[64 * 64];
  __shared__ float cl[64];
  const int tid = threadIdx.x;
  const float h = kDT;

  // M = -2 W W^T ; c = -2 b W^T
  for (int e = tid; e < 4096; e += 256) {
    const int i = e >> 6, j = e & 63;
    float acc = 0.f;
    for (int k = 0; k < 64; ++k) acc += W[i * 64 + k] * W[j * 64 + k];
    Ml[e] = -2.f * acc;
  }
  if (tid < 64) {
    float acc = 0.f;
    for (int k = 0; k < 64; ++k) acc += bias[k] * W[tid * 64 + k];
    cl[tid] = -2.f * acc;
  }
  __syncthreads();

  // A = M^2
  for (int e = tid; e < 4096; e += 256) {
    const int i = e >> 6, j = e & 63;
    float acc = 0.f;
    for (int k = 0; k < 64; ++k) acc += Ml[i * 64 + k] * Ml[k * 64 + j];
    Al[e] = acc;
  }
  __syncthreads();

  // B = M^3
  for (int e = tid; e < 4096; e += 256) {
    const int i = e >> 6, j = e & 63;
    float acc = 0.f;
    for (int k = 0; k < 64; ++k) acc += Al[i * 64 + k] * Ml[k * 64 + j];
    Bl[e] = acc;
  }
  __syncthreads();

  // B <- G (elementwise, in place)
  for (int e = tid; e < 4096; e += 256) {
    const int i = e >> 6, j = e & 63;
    Bl[e] = (i == j ? 1.f : 0.f) + (h * 0.5f) * Ml[e] +
            (h * h / 6.f) * Al[e] + (h * h * h / 24.f) * Bl[e];
  }
  __syncthreads();

  // A <- M*G ; s = h * c * G
  for (int e = tid; e < 4096; e += 256) {
    const int i = e >> 6, j = e & 63;
    float acc = 0.f;
    for (int k = 0; k < 64; ++k) acc += Ml[i * 64 + k] * Bl[k * 64 + j];
    Al[e] = acc;
  }
  if (tid < 64) {
    float acc = 0.f;
    for (int k = 0; k < 64; ++k) acc += cl[k] * Bl[k * 64 + tid];
    sOut[tid] = h * acc;
    cOut[tid] = cl[tid];
  }
  __syncthreads();

  // A <- R = I + h*(M*G)
  for (int e = tid; e < 4096; e += 256) {
    const int i = e >> 6, j = e & 63;
    Al[e] = (i == j ? 1.f : 0.f) + h * Al[e];
  }
  __syncthreads();

  // Transposed f16 copies: MT[n*64+k] = M[k][n]
  for (int e = tid; e < 4096; e += 256) {
    const int n = e >> 6, k = e & 63;
    MT[e] = (_Float16)Ml[k * 64 + n];
    RT[e] = (_Float16)Al[k * 64 + n];
  }
}

// ---------------------------------------------------------------------------
// Main: each wave owns 16 samples (rows). Per step:
//   v = xM + c, u = xR + s  via 16x v_wmma_f32_16x16x32_f16 (weight-stationary),
//   LDS round trip to rotate C/D f32 -> A-operand layout, freeze test, select.
// A-operand layout (ISA 7.12.2): lane l holds row l%16; 8-element K runs at
//   {0,16}+32q (lanes 0-15) or {8,24}+32q (lanes 16-31).
// B-operand: lane l holds column 16n+l%16, 8-element K runs at 32q + 8*(l>=16) + {0,16} (same as A).
// C/D: VGPR r -> row r + 8*(l>=16), lane -> col 16n + l%16.
// ---------------------------------------------------------------------------
__global__ void __launch_bounds__(256) hop_main(
    const float* __restrict__ x0,
    const _Float16* __restrict__ MT,
    const _Float16* __restrict__ RT,
    const float* __restrict__ cvec,
    const float* __restrict__ svec,
    float* __restrict__ out) {
  __shared__ float lbuf[8 * 2048];  // per wave: u[16][64] + v[16][64] f32
  const int tid  = threadIdx.x;
  const int wave = tid >> 5;
  const int lane = tid & 31;
  const int ln   = lane & 15;
  const int hi   = lane >> 4;
  const int rb0  = hi ? 8 : 0;    // first 8-run base within a 32-K chunk
  const int rb1  = hi ? 24 : 16;  // second 8-run base

  float* ul = lbuf + wave * 2048;
  float* vl = ul + 1024;
  const int    rowBase = blockIdx.x * 128 + wave * 16;
  const size_t row     = (size_t)(rowBase + ln) * 64;

  // Master x (fp32) kept directly in A-operand element order.
  float xr[32];
  {
    const float* p = x0 + row;
#pragma unroll
    for (int q = 0; q < 2; ++q) {
      float4 f0 = *(const float4*)(p + 32 * q + rb0);
      float4 f1 = *(const float4*)(p + 32 * q + rb0 + 4);
      float4 f2 = *(const float4*)(p + 32 * q + rb1);
      float4 f3 = *(const float4*)(p + 32 * q + rb1 + 4);
      float* d = xr + 16 * q;
      d[0] = f0.x; d[1] = f0.y; d[2]  = f0.z; d[3]  = f0.w;
      d[4] = f1.x; d[5] = f1.y; d[6]  = f1.z; d[7]  = f1.w;
      d[8] = f2.x; d[9] = f2.y; d[10] = f2.z; d[11] = f2.w;
      d[12] = f3.x; d[13] = f3.y; d[14] = f3.z; d[15] = f3.w;
    }
  }

  // Weight-stationary B operands: M and R, 2 K-chunks x 4 column tiles each.
  union UB { v16h v; uint4 u4[2]; };
  UB Mb[2][4], Rb[2][4];
#pragma unroll
  for (int q = 0; q < 2; ++q) {
#pragma unroll
    for (int n = 0; n < 4; ++n) {
      const int col  = 16 * n + ln;
      const int koff = 32 * q + 8 * hi;
      const uint4* pm = (const uint4*)(MT + col * 64 + koff);
      const uint4* pr = (const uint4*)(RT + col * 64 + koff);
      Mb[q][n].u4[0] = pm[0]; Mb[q][n].u4[1] = pm[2];
      Rb[q][n].u4[0] = pr[0]; Rb[q][n].u4[1] = pr[2];
    }
  }
  float cval[4], sval[4];
#pragma unroll
  for (int n = 0; n < 4; ++n) {
    cval[n] = cvec[16 * n + ln];
    sval[n] = svec[16 * n + ln];
  }

  for (int t = 0; t < kSteps; ++t) {
    UB a0, a1;
#pragma unroll
    for (int j = 0; j < 16; ++j) {
      a0.v[j] = (_Float16)xr[j];
      a1.v[j] = (_Float16)xr[16 + j];
    }

#pragma unroll
    for (int n = 0; n < 4; ++n) {
      v8f vac = {cval[n], cval[n], cval[n], cval[n], cval[n], cval[n], cval[n], cval[n]};
      v8f uac = {sval[n], sval[n], sval[n], sval[n], sval[n], sval[n], sval[n], sval[n]};
      vac = __builtin_amdgcn_wmma_f32_16x16x32_f16(false, a0.v, false, Mb[0][n].v, (short)0, vac, false, false);
      vac = __builtin_amdgcn_wmma_f32_16x16x32_f16(false, a1.v, false, Mb[1][n].v, (short)0, vac, false, false);
      uac = __builtin_amdgcn_wmma_f32_16x16x32_f16(false, a0.v, false, Rb[0][n].v, (short)0, uac, false, false);
      uac = __builtin_amdgcn_wmma_f32_16x16x32_f16(false, a1.v, false, Rb[1][n].v, (short)0, uac, false, false);
      const int ff = 16 * n + ln;
#pragma unroll
      for (int r = 0; r < 8; ++r) {
        const int mm = r + 8 * hi;            // C/D row -> sample
        ul[mm * 64 + ff] = uac[r];
        vl[mm * 64 + ff] = vac[r];
      }
    }
    __syncthreads();

    // Re-read in A-operand run layout; fold |v| max over this lane's half-row.
    float ur[32];
    float mx = 0.f;
#pragma unroll
    for (int q = 0; q < 2; ++q) {
      const float* up = ul + ln * 64 + 32 * q;
      const float* vp = vl + ln * 64 + 32 * q;
      float4 f0 = *(const float4*)(up + rb0);
      float4 f1 = *(const float4*)(up + rb0 + 4);
      float4 f2 = *(const float4*)(up + rb1);
      float4 f3 = *(const float4*)(up + rb1 + 4);
      float* d = ur + 16 * q;
      d[0] = f0.x; d[1] = f0.y; d[2]  = f0.z; d[3]  = f0.w;
      d[4] = f1.x; d[5] = f1.y; d[6]  = f1.z; d[7]  = f1.w;
      d[8] = f2.x; d[9] = f2.y; d[10] = f2.z; d[11] = f2.w;
      d[12] = f3.x; d[13] = f3.y; d[14] = f3.z; d[15] = f3.w;
      float4 g0 = *(const float4*)(vp + rb0);
      float4 g1 = *(const float4*)(vp + rb0 + 4);
      float4 g2 = *(const float4*)(vp + rb1);
      float4 g3 = *(const float4*)(vp + rb1 + 4);
      mx = fmaxf(mx, fabsf(g0.x)); mx = fmaxf(mx, fabsf(g0.y));
      mx = fmaxf(mx, fabsf(g0.z)); mx = fmaxf(mx, fabsf(g0.w));
      mx = fmaxf(mx, fabsf(g1.x)); mx = fmaxf(mx, fabsf(g1.y));
      mx = fmaxf(mx, fabsf(g1.z)); mx = fmaxf(mx, fabsf(g1.w));
      mx = fmaxf(mx, fabsf(g2.x)); mx = fmaxf(mx, fabsf(g2.y));
      mx = fmaxf(mx, fabsf(g2.z)); mx = fmaxf(mx, fabsf(g2.w));
      mx = fmaxf(mx, fabsf(g3.x)); mx = fmaxf(mx, fabsf(g3.y));
      mx = fmaxf(mx, fabsf(g3.z)); mx = fmaxf(mx, fabsf(g3.w));
    }
    // lanes l and l+16 hold the two halves of the same row
    mx = fmaxf(mx, __shfl_xor(mx, 16, 32));
    const bool act = (mx >= kTOL);
#pragma unroll
    for (int j = 0; j < 32; ++j) xr[j] = act ? ur[j] : xr[j];
    __syncthreads();
  }

  // Scatter result back (same run layout as the load).
  {
    float* p = out + row;
#pragma unroll
    for (int q = 0; q < 2; ++q) {
      const float* d = xr + 16 * q;
      *(float4*)(p + 32 * q + rb0)     = make_float4(d[0], d[1], d[2], d[3]);
      *(float4*)(p + 32 * q + rb0 + 4) = make_float4(d[4], d[5], d[6], d[7]);
      *(float4*)(p + 32 * q + rb1)     = make_float4(d[8], d[9], d[10], d[11]);
      *(float4*)(p + 32 * q + rb1 + 4) = make_float4(d[12], d[13], d[14], d[15]);
    }
  }
}

extern "C" void kernel_launch(void* const* d_in, const int* in_sizes, int n_in,
                              void* d_out, int out_size, void* d_ws, size_t ws_size,
                              hipStream_t stream) {
  const float* x0 = (const float*)d_in[0];
  const float* W  = (const float*)d_in[1];
  const float* b  = (const float*)d_in[2];

  _Float16* MT = (_Float16*)d_ws;                 // 64*64 f16
  _Float16* RT = MT + 64 * 64;                    // 64*64 f16
  float* cv = (float*)((char*)d_ws + 2 * 64 * 64 * sizeof(_Float16));
  float* sv = cv + 64;
  float* out = (float*)d_out;

  const int B = in_sizes[0] / 64;                 // 262144 samples

  hop_setup<<<1, 256, 0, stream>>>(W, b, MT, RT, cv, sv);
  hop_main<<<B / 128, 256, 0, stream>>>(x0, MT, RT, cv, sv, out);
}
